// CausalWanSelfAttention_78211354460596
// MI455X (gfx1250) — matrix operand order repaired and checked by reading, not compiled, not run on hardware
//
#include <hip/hip_runtime.h>
#include <hip/hip_bf16.h>
#include <cstdint>

// ---------------------------------------------------------------------------
// Problem constants (from reference)
// ---------------------------------------------------------------------------
constexpr int S    = 6240;
constexpr int DIM  = 1536;
constexpr int H    = 12;
constexpr int D    = 128;
constexpr int L    = 1560;
constexpr int F    = 4;       // S / L
constexpr int HD   = H * D;   // 1536
constexpr int E3   = 3 * HD;  // 4608
constexpr float SCALE = 0.08838834764831845f;  // 1/sqrt(128)
constexpr float EPSV  = 1e-6f;
constexpr float NEGV  = -1e9f;

typedef __bf16 bf16_t;
typedef __attribute__((ext_vector_type(16))) __bf16 v16bf;
typedef __attribute__((ext_vector_type(8)))  __bf16 v8bf;
typedef __attribute__((ext_vector_type(8)))  float  v8f;

// ---------------------------------------------------------------------------
// Kernel 0: fp32 -> bf16 conversion
// ---------------------------------------------------------------------------
__global__ __launch_bounds__(256)
void cvt_f32_bf16(const float* __restrict__ in, bf16_t* __restrict__ out, int n)
{
    int i = blockIdx.x * 256 + threadIdx.x;
    if (i < n) out[i] = (bf16_t)in[i];
}

// ---------------------------------------------------------------------------
// WMMA GEMM:  C[M,N] = A[M,K] * B[N,K]^T + bias[N]    (A,B bf16, C fp32)
// Block: 256 threads = 8 waves arranged 2(M) x 4(N); block tile 64(M) x 128(N)
// Each wave computes a 32x32 patch = 2x2 v_wmma_f32_16x16x32_bf16 tiles.
// ---------------------------------------------------------------------------
__global__ __launch_bounds__(256)
void gemm_bf16_wmma(const bf16_t* __restrict__ A, const bf16_t* __restrict__ Bm,
                    const float* __restrict__ bias, float* __restrict__ Cout,
                    int M, int N, int K)
{
    const int lane = threadIdx.x & 31;
    const int wid  = threadIdx.x >> 5;
    const int half = lane >> 4;
    const int l16  = lane & 15;
    const int wm   = wid & 1;        // 0..1
    const int wn   = wid >> 1;       // 0..3
    const int mbase = blockIdx.y * 64 + wm * 32;
    const int nbase = blockIdx.x * 128 + wn * 32;

    v8f acc[2][2] = {};

    int arow[2];
#pragma unroll
    for (int mt = 0; mt < 2; ++mt) {
        int r = mbase + mt * 16 + l16;      // A-frag: M = lane%16
        arow[mt] = (r < M) ? r : (M - 1);   // clamp (stores guarded)
    }
    int ncol[2];
#pragma unroll
    for (int nt = 0; nt < 2; ++nt) ncol[nt] = nbase + nt * 16 + l16;  // B-frag: N = lane%16

    for (int k0 = 0; k0 < K; k0 += 32) {
        v16bf a[2], b[2];
#pragma unroll
        for (int mt = 0; mt < 2; ++mt) {
            // A 16x32 layout: lane half0 holds K {0..15}, half1 {16..31}
            const bf16_t* ap = A + (size_t)arow[mt] * K + k0;
            v8bf lo = *reinterpret_cast<const v8bf*>(ap + 16 * half);
            v8bf hi = *reinterpret_cast<const v8bf*>(ap + 16 * half + 8);
#pragma unroll
            for (int j = 0; j < 8; ++j) { a[mt][j] = lo[j]; a[mt][8 + j] = hi[j]; }
        }
#pragma unroll
        for (int nt = 0; nt < 2; ++nt) {
            // B 32x16 layout: b[j] = B[k = j + 16*half][n = lane%16], contiguous K
            b[nt] = *reinterpret_cast<const v16bf*>(Bm + (size_t)ncol[nt] * K + k0 + 16 * half);
        }
#pragma unroll
        for (int mt = 0; mt < 2; ++mt)
#pragma unroll
            for (int nt = 0; nt < 2; ++nt)
                acc[mt][nt] = __builtin_amdgcn_wmma_f32_16x16x32_bf16(
                    false, a[mt], false, b[nt], (short)0, acc[mt][nt], false, false);
    }

    // ---- epilogue: bias loaded once per wave; branch-free fast path
    float bias_v[2];
#pragma unroll
    for (int nt = 0; nt < 2; ++nt) bias_v[nt] = bias[ncol[nt]];

    if (mbase + 32 <= M) {
        // entire 32-row wave patch in range: straight-line stores
#pragma unroll
        for (int mt = 0; mt < 2; ++mt) {
            const int row0 = mbase + mt * 16 + 8 * half;   // C/D: M = r + 8*half
#pragma unroll
            for (int nt = 0; nt < 2; ++nt) {
                float* p = Cout + (size_t)row0 * N + ncol[nt];
#pragma unroll
                for (int r = 0; r < 8; ++r) {
                    *p = acc[mt][nt][r] + bias_v[nt];
                    p += N;
                }
            }
        }
    } else {
#pragma unroll
        for (int mt = 0; mt < 2; ++mt)
#pragma unroll
            for (int r = 0; r < 8; ++r) {
                const int row = mbase + mt * 16 + r + 8 * half;
                if (row < M) {
                    float* p = Cout + (size_t)row * N;
#pragma unroll
                    for (int nt = 0; nt < 2; ++nt)
                        p[ncol[nt]] = acc[mt][nt][r] + bias_v[nt];
                }
            }
    }
}

// ---------------------------------------------------------------------------
// Kernel: per-token RMSNorm(q,k) + RoPE(q,k) + repack to bf16
//   q -> [H][S][D] bf16, pre-scaled by 1/sqrt(D) (folded softmax scale)
//   k -> [H][S][D] bf16
//   v -> [H][D][S] bf16 (transposed: PV B-fragments contiguous over keys)
// One block (256 threads) per token.
// ---------------------------------------------------------------------------
__global__ __launch_bounds__(256)
void normrope_pack(const float* __restrict__ qkv, const float* __restrict__ cosb,
                   const float* __restrict__ sinb, const float* __restrict__ gq,
                   const float* __restrict__ gk,
                   bf16_t* __restrict__ qb, bf16_t* __restrict__ kb,
                   bf16_t* __restrict__ vt)
{
    const int s   = blockIdx.x;
    const int tid = threadIdx.x;
    __shared__ float sQ[HD];
    __shared__ float sK[HD];
    __shared__ float rq8[8], rk8[8];
    __shared__ float sScale[2];

    const float* row = qkv + (size_t)s * E3;
    float ssq = 0.f, ssk = 0.f;
    for (int e = tid; e < HD; e += 256) {
        float q = row[e], k = row[HD + e];
        sQ[e] = q; sK[e] = k;
        ssq += q * q; ssk += k * k;
    }
#pragma unroll
    for (int m = 1; m < 32; m <<= 1) {
        ssq += __shfl_xor(ssq, m, 32);
        ssk += __shfl_xor(ssk, m, 32);
    }
    if ((tid & 31) == 0) { rq8[tid >> 5] = ssq; rk8[tid >> 5] = ssk; }
    __syncthreads();
    if (tid == 0) {
        float a = 0.f, b = 0.f;
#pragma unroll
        for (int i = 0; i < 8; ++i) { a += rq8[i]; b += rk8[i]; }
        sScale[0] = rsqrtf(a / (float)HD + EPSV) * SCALE;  // fold softmax scale into q
        sScale[1] = rsqrtf(b / (float)HD + EPSV);
    }
    __syncthreads();
    const float rq = sScale[0], rk = sScale[1];

    // RoPE over 768 (h,j) pairs: dims (2j, 2j+1) of head h
    for (int i = tid; i < H * (D / 2); i += 256) {
        int h = i >> 6, j = i & 63;
        float c  = cosb[s * 64 + j];
        float sn = sinb[s * 64 + j];
        float q1 = sQ[2 * i]     * rq * gq[2 * i];
        float q2 = sQ[2 * i + 1] * rq * gq[2 * i + 1];
        float k1 = sK[2 * i]     * rk * gk[2 * i];
        float k2 = sK[2 * i + 1] * rk * gk[2 * i + 1];
        size_t base = ((size_t)h * S + s) * D;
        qb[base + 2 * j]     = (bf16_t)(q1 * c - q2 * sn);
        qb[base + 2 * j + 1] = (bf16_t)(q1 * sn + q2 * c);
        kb[base + 2 * j]     = (bf16_t)(k1 * c - k2 * sn);
        kb[base + 2 * j + 1] = (bf16_t)(k1 * sn + k2 * c);
    }
    // V transpose -> [H][D][S]
    for (int e = tid; e < HD; e += 256) {
        int h = e >> 7, d = e & 127;
        vt[((size_t)h * D + d) * S + s] = (bf16_t)row[2 * HD + e];
    }
}

// ---------------------------------------------------------------------------
// Attention kernel: one wave per (query-16-tile, frame, head).
// Frame f attends to frames [f-1, f] (frame 0: just frame 0), no intra mask.
// Online softmax over key steps of 32; scores: 8 WMMAs (K=D=128 in 4 chunks,
// two 16-key tiles); probs go C-layout -> LDS -> A-layout; PV: 8 WMMAs.
// Q is pre-scaled by 1/sqrt(D).
// ---------------------------------------------------------------------------
__global__ __launch_bounds__(32)
void attn_kernel(const bf16_t* __restrict__ qb, const bf16_t* __restrict__ kb,
                 const bf16_t* __restrict__ vt, bf16_t* __restrict__ attn)
{
    const int qt   = blockIdx.x;   // 0..97 (98 tiles of 16 queries; L=1560)
    const int f    = blockIdx.y;   // 0..3
    const int h    = blockIdx.z;   // 0..11
    const int lane = threadIdx.x;
    const int half = lane >> 4;
    const int l16  = lane & 15;

    __shared__ float sP[16][36];   // stride 36: conflict-free on 64 banks

    // ---- preload Q fragments (A-matrix 16x32, 4 K-chunks over D=128)
    int qrow   = qt * 16 + l16;                        // A-frag M = lane%16
    int qrow_c = (qrow < L) ? qrow : (L - 1);
    const bf16_t* qptr = qb + ((size_t)h * S + (size_t)f * L + qrow_c) * D;
    v16bf qf[4];
#pragma unroll
    for (int kc = 0; kc < 4; ++kc) {
        v8bf lo = *reinterpret_cast<const v8bf*>(qptr + kc * 32 + 16 * half);
        v8bf hi = *reinterpret_cast<const v8bf*>(qptr + kc * 32 + 16 * half + 8);
#pragma unroll
        for (int j = 0; j < 8; ++j) { qf[kc][j] = lo[j]; qf[kc][8 + j] = hi[j]; }
    }

    v8f O[8] = {};
    float m[8], lsum[8];
#pragma unroll
    for (int r = 0; r < 8; ++r) { m[r] = -3.0e38f; lsum[r] = 0.f; }

    const int kseq0  = (f == 0) ? 0 : (f - 1) * L;     // window base in sequence
    const int KV     = (f == 0) ? L : 2 * L;           // valid keys in window
    const int nsteps = (KV + 31) / 32;                 // 49 or 98

    for (int step = 0; step < nsteps; ++step) {
        // ---- scores for 32 keys (two 16x16 C tiles)
        v8f s0 = {}, s1 = {};
        int kp0 = step * 32 + l16;                     // B-frag N = key = lane%16
        int kp1 = kp0 + 16;
        int ks0 = kseq0 + kp0; if (ks0 > S - 1) ks0 = S - 1;
        int ks1 = kseq0 + kp1; if (ks1 > S - 1) ks1 = S - 1;
        const bf16_t* kp0p = kb + ((size_t)h * S + ks0) * D;
        const bf16_t* kp1p = kb + ((size_t)h * S + ks1) * D;
#pragma unroll
        for (int kc = 0; kc < 4; ++kc) {
            v16bf b0 = *reinterpret_cast<const v16bf*>(kp0p + kc * 32 + 16 * half);
            v16bf b1 = *reinterpret_cast<const v16bf*>(kp1p + kc * 32 + 16 * half);
            s0 = __builtin_amdgcn_wmma_f32_16x16x32_bf16(false, qf[kc], false, b0,
                                                         (short)0, s0, false, false);
            s1 = __builtin_amdgcn_wmma_f32_16x16x32_bf16(false, qf[kc], false, b1,
                                                         (short)0, s1, false, false);
        }

        const bool v0 = (kp0 < KV);
        const bool v1 = (kp1 < KV);

        // ---- online softmax (rows live in 16-lane halves; reduce via shfl_xor)
        float alpha[8], p0[8], p1[8];
#pragma unroll
        for (int r = 0; r < 8; ++r) {
            float a0 = v0 ? s0[r] : NEGV;              // Q pre-scaled by 1/sqrt(D)
            float a1 = v1 ? s1[r] : NEGV;
            float mx = fmaxf(a0, a1);
            mx = fmaxf(mx, __shfl_xor(mx, 1, 32));
            mx = fmaxf(mx, __shfl_xor(mx, 2, 32));
            mx = fmaxf(mx, __shfl_xor(mx, 4, 32));
            mx = fmaxf(mx, __shfl_xor(mx, 8, 32));
            float mn = fmaxf(m[r], mx);
            float al = __expf(m[r] - mn);
            float e0 = __expf(a0 - mn);
            float e1 = __expf(a1 - mn);
            float rs = e0 + e1;
            rs += __shfl_xor(rs, 1, 32);
            rs += __shfl_xor(rs, 2, 32);
            rs += __shfl_xor(rs, 4, 32);
            rs += __shfl_xor(rs, 8, 32);
            lsum[r] = lsum[r] * al + rs;
            m[r] = mn;
            alpha[r] = al;
            p0[r] = e0; p1[r] = e1;
        }
#pragma unroll
        for (int t = 0; t < 8; ++t)
#pragma unroll
            for (int r = 0; r < 8; ++r) O[t][r] *= alpha[r];

        // ---- probs: C layout -> LDS -> A layout
#pragma unroll
        for (int r = 0; r < 8; ++r) {
            sP[r + 8 * half][l16]      = p0[r];
            sP[r + 8 * half][16 + l16] = p1[r];
        }
        __syncthreads();
        v16bf pa;
#pragma unroll
        for (int j = 0; j < 8; ++j) {
            pa[j]     = (bf16_t)sP[l16][16 * half + j];
            pa[8 + j] = (bf16_t)sP[l16][16 * half + 8 + j];
        }
        __syncthreads();

        // ---- PV: O[t] += P(16x32) * V(32x16) for 8 d-tiles
        int kbse = kseq0 + step * 32 + 16 * half;      // chunk base (contig keys)
        if (kbse > S - 16) kbse = S - 16;              // only masked keys clamp
#pragma unroll
        for (int t = 0; t < 8; ++t) {
            const bf16_t* vp = vt + ((size_t)h * D + t * 16 + l16) * S + kbse;
            v16bf bv = *reinterpret_cast<const v16bf*>(vp);
            O[t] = __builtin_amdgcn_wmma_f32_16x16x32_bf16(false, pa, false, bv,
                                                           (short)0, O[t], false, false);
        }
    }

    // ---- finalize and store (attn row-major [S][H*D], bf16)
    if (qt * 16 + 16 <= L) {
#pragma unroll
        for (int r = 0; r < 8; ++r) {
            const int qr = qt * 16 + r + 8 * half;     // C/D: M = r + 8*half
            const float inv = 1.f / lsum[r];
            bf16_t* p = attn + (size_t)(f * L + qr) * HD + (size_t)h * D + l16;
#pragma unroll
            for (int t = 0; t < 8; ++t)
                p[t * 16] = (bf16_t)(O[t][r] * inv);
        }
    } else {
#pragma unroll
        for (int r = 0; r < 8; ++r) {
            const int qr = qt * 16 + r + 8 * half;
            if (qr < L) {
                const float inv = 1.f / lsum[r];
                bf16_t* p = attn + (size_t)(f * L + qr) * HD + (size_t)h * D + l16;
#pragma unroll
                for (int t = 0; t < 8; ++t)
                    p[t * 16] = (bf16_t)(O[t][r] * inv);
            }
        }
    }
}

// ---------------------------------------------------------------------------
// Host launcher
// ---------------------------------------------------------------------------
extern "C" void kernel_launch(void* const* d_in, const int* in_sizes, int n_in,
                              void* d_out, int out_size, void* d_ws, size_t ws_size,
                              hipStream_t stream)
{
    (void)in_sizes; (void)n_in; (void)out_size; (void)ws_size;
    const float* hidden = (const float*)d_in[0];
    const float* fcos   = (const float*)d_in[1];
    const float* fsin   = (const float*)d_in[2];
    const float* Wqkv   = (const float*)d_in[3];
    const float* bqkv   = (const float*)d_in[4];
    const float* gq     = (const float*)d_in[5];
    const float* gk     = (const float*)d_in[6];
    const float* Wout   = (const float*)d_in[7];
    const float* bout   = (const float*)d_in[8];
    float* out = (float*)d_out;

    char* ws = (char*)d_ws;
    size_t off = 0;
    auto carve = [&](size_t bytes) -> char* {
        char* p = ws + off;
        off += (bytes + 255) & ~(size_t)255;
        return p;
    };
    bf16_t* hid_bf  = (bf16_t*)carve((size_t)S * DIM * 2);   // 19.2 MB
    bf16_t* wqkv_bf = (bf16_t*)carve((size_t)E3 * DIM * 2);  // 14.2 MB
    bf16_t* wout_bf = (bf16_t*)carve((size_t)DIM * HD * 2);  //  4.7 MB
    float*  qkv_f   = (float*)carve((size_t)S * E3 * 4);     // 115  MB
    bf16_t* q_bf    = (bf16_t*)carve((size_t)H * S * D * 2); // 19.2 MB
    bf16_t* k_bf    = (bf16_t*)carve((size_t)H * S * D * 2); // 19.2 MB
    bf16_t* vt_bf   = (bf16_t*)carve((size_t)H * D * S * 2); // 19.2 MB
    bf16_t* attn_bf = (bf16_t*)carve((size_t)S * HD * 2);    // 19.2 MB

    // 1) fp32 -> bf16 conversions
    {
        int n1 = S * DIM, n2 = E3 * DIM, n3 = DIM * HD;
        cvt_f32_bf16<<<(n1 + 255) / 256, 256, 0, stream>>>(hidden, hid_bf, n1);
        cvt_f32_bf16<<<(n2 + 255) / 256, 256, 0, stream>>>(Wqkv, wqkv_bf, n2);
        cvt_f32_bf16<<<(n3 + 255) / 256, 256, 0, stream>>>(Wout, wout_bf, n3);
    }
    // 2) QKV GEMM: [S,DIM] x [E3,DIM]^T + b_qkv -> qkv fp32
    {
        dim3 grid(E3 / 128, (S + 63) / 64);
        gemm_bf16_wmma<<<grid, 256, 0, stream>>>(hid_bf, wqkv_bf, bqkv, qkv_f,
                                                 S, E3, DIM);
    }
    // 3) RMSNorm + RoPE + pack
    normrope_pack<<<S, 256, 0, stream>>>(qkv_f, fcos, fsin, gq, gk,
                                         q_bf, k_bf, vt_bf);
    // 4) windowed attention
    {
        dim3 grid((L + 15) / 16, F, H);   // 98 x 4 x 12 waves
        attn_kernel<<<grid, 32, 0, stream>>>(q_bf, k_bf, vt_bf, attn_bf);
    }
    // 5) output GEMM: [S,HD] x [DIM,HD]^T + b_out -> out fp32
    {
        dim3 grid(DIM / 128, (S + 63) / 64);
        gemm_bf16_wmma<<<grid, 256, 0, stream>>>(attn_bf, wout_bf, bout, out,
                                                 S, DIM, HD);
    }
}
